// DeepseekV3MoE_40492951666927
// MI455X (gfx1250) — matrix operand order repaired and checked by reading, not compiled, not run on hardware
//
#include <hip/hip_runtime.h>
#include <hip/hip_bf16.h>
#include <cstdint>
#include <cstddef>

// ---------------- problem constants (from reference) ----------------
#define T_    4096        // B*S tokens
#define D_    2048        // hidden
#define I_    1024        // moe intermediate
#define E_    32          // routed experts
#define K_    4           // experts per token
#define NG_   8           // n_group
#define TKG_  3           // topk_group
#define ISH_  2048        // I * NSH (shared intermediate)
#define C_    160         // capacity = ceil(1.25*T/E)
#define EC_   5120        // E * C
#define SCALE_ 2.5f

typedef __attribute__((ext_vector_type(16))) _Float16 v16h;
typedef __attribute__((ext_vector_type(8)))  float    v8f;

// ---------------- WMMA helpers (CDNA5 gfx1250, wave32) ----------------
__device__ __forceinline__ v8f wmma32(v16h a, v16h b, v8f c) {
  // v_wmma_f32_16x16x32_f16  D = A(16x32) * B(32x16) + C
  return __builtin_amdgcn_wmma_f32_16x16x32_f16(false, a, false, b, (short)0, c,
                                                false, false);
}

// A fragment (16x32 f16): lane r = row; half h -> K = h*16..h*16+15 contiguous
// rowp = &tile[row][kkBase]   (row stride passed in as halfs)
__device__ __forceinline__ v16h lds_fragA(const _Float16* rowp, int kOff) {
  union { uint4 q[2]; v16h v; } u;
  u.q[0] = *(const uint4*)(rowp + 2 * kOff);
  u.q[1] = *(const uint4*)(rowp + 2 * kOff + 8);
  return u.v;
}

// B fragment (32x16 f16): lane r = column; half h -> K = h*16..h*16+15 contiguous.
// rowp = &tileT[col][kkBase]  (K-major transposed tile)
__device__ __forceinline__ v16h lds_fragB(const _Float16* rowp, int half) {
  union { uint4 q[2]; v16h v; } u;
  const _Float16* p = rowp + half * 16;
  u.q[0] = *(const uint4*)(p);
  u.q[1] = *(const uint4*)(p + 8);
  return u.v;
}

// B fragment straight from an [N][K]-major fp32 global matrix row (contiguous K).
__device__ __forceinline__ v16h glb_fragB_f32(const float* rowp, int half) {
  const float* p = rowp + half * 16;
  float4 a = *(const float4*)(p + 0);
  float4 b = *(const float4*)(p + 4);
  float4 c = *(const float4*)(p + 8);
  float4 d = *(const float4*)(p + 12);
  v16h v;
  v[0]  = (_Float16)a.x; v[1]  = (_Float16)a.y; v[2]  = (_Float16)a.z; v[3]  = (_Float16)a.w;
  v[4]  = (_Float16)b.x; v[5]  = (_Float16)b.y; v[6]  = (_Float16)b.z; v[7]  = (_Float16)b.w;
  v[8]  = (_Float16)c.x; v[9]  = (_Float16)c.y; v[10] = (_Float16)c.z; v[11] = (_Float16)c.w;
  v[12] = (_Float16)d.x; v[13] = (_Float16)d.y; v[14] = (_Float16)d.z; v[15] = (_Float16)d.w;
  return v;
}

// pack 8 floats -> 8 f16 -> one 16B LDS store
__device__ __forceinline__ void store8h(_Float16* dst, float4 a, float4 b) {
  union { _Float16 h[8]; uint4 q; } u;
  u.h[0] = (_Float16)a.x; u.h[1] = (_Float16)a.y; u.h[2] = (_Float16)a.z; u.h[3] = (_Float16)a.w;
  u.h[4] = (_Float16)b.x; u.h[5] = (_Float16)b.y; u.h[6] = (_Float16)b.z; u.h[7] = (_Float16)b.w;
  *(uint4*)dst = u.q;
}

__device__ __forceinline__ float silu_f(float g) {
  return g / (1.0f + __expf(-g));
}

// =====================================================================
// 1) Router scores: one wave per token, lane = expert (E_ == 32 == wave32)
// =====================================================================
__global__ __launch_bounds__(32)
void router_scores_kernel(const float* __restrict__ x,
                          const float* __restrict__ rw,
                          float* __restrict__ scores) {
  const int t = blockIdx.x;
  const int e = threadIdx.x;
  const float* xr = x + (size_t)t * D_;
  const float* wr = rw + (size_t)e * D_;
  float s = 0.0f;
  for (int d = 0; d < D_; d += 4) {
    float4 a = *(const float4*)(xr + d);
    float4 b = *(const float4*)(wr + d);
    s += a.x * b.x + a.y * b.y + a.z * b.z + a.w * b.w;
  }
  scores[t * E_ + e] = 1.0f / (1.0f + __expf(-s));   // sigmoid
}

// =====================================================================
// 2) Grouped top-k routing (one thread per token, stable tie-breaks)
// =====================================================================
__global__ __launch_bounds__(256)
void topk_kernel(const float* __restrict__ scores,
                 const float* __restrict__ e_bias,
                 int* __restrict__ topk_idx,
                 float* __restrict__ topk_w) {
  const int t = blockIdx.x * blockDim.x + threadIdx.x;
  if (t >= T_) return;
  float s0[E_], sc[E_];
  for (int e = 0; e < E_; ++e) {
    s0[e] = scores[t * E_ + e];
    sc[e] = s0[e] + e_bias[e];
  }
  float gs[NG_];
  for (int g = 0; g < NG_; ++g) {
    float m1 = -1e30f, m2 = -1e30f;
    for (int j = 0; j < E_ / NG_; ++j) {
      float v = sc[g * (E_ / NG_) + j];
      if (v > m1) { m2 = m1; m1 = v; } else if (v > m2) { m2 = v; }
    }
    gs[g] = m1 + m2;
  }
  bool gsel[NG_];
  for (int g = 0; g < NG_; ++g) gsel[g] = false;
  for (int it = 0; it < TKG_; ++it) {
    float best = -1e30f; int bi = 0;
    for (int g = 0; g < NG_; ++g)
      if (!gsel[g] && gs[g] > best) { best = gs[g]; bi = g; }
    gsel[bi] = true;
  }
  float masked[E_];
  for (int e = 0; e < E_; ++e)
    masked[e] = gsel[e / (E_ / NG_)] ? sc[e] : 0.0f;
  int idx[K_]; float w[K_]; float wsum = 0.0f;
  for (int kk = 0; kk < K_; ++kk) {
    float best = -1e30f; int bi = 0;
    for (int e = 0; e < E_; ++e)
      if (masked[e] > best) { best = masked[e]; bi = e; }
    masked[bi] = -1e30f;
    idx[kk] = bi;
    w[kk] = s0[bi];
    wsum += w[kk];
  }
  const float inv = SCALE_ / (wsum + 1e-20f);
  for (int kk = 0; kk < K_; ++kk) {
    topk_idx[t * K_ + kk] = idx[kk];
    topk_w[t * K_ + kk]   = w[kk] * inv;
  }
}

// =====================================================================
// 3) Dispatch: deterministic (stable) per-expert ranks, capacity drop
// =====================================================================
__global__ __launch_bounds__(256)
void count_kernel(const int* __restrict__ topk_idx, int* __restrict__ bCnt) {
  __shared__ int cnt[E_];
  if (threadIdx.x < E_) cnt[threadIdx.x] = 0;
  __syncthreads();
  const int i = blockIdx.x * 256 + threadIdx.x;
  atomicAdd(&cnt[topk_idx[i]], 1);
  __syncthreads();
  if (threadIdx.x < E_) bCnt[blockIdx.x * E_ + threadIdx.x] = cnt[threadIdx.x];
}

__global__ __launch_bounds__(32)
void scan_kernel(const int* __restrict__ bCnt, int* __restrict__ bOff) {
  const int e = threadIdx.x;
  int run = 0;
  for (int b = 0; b < 64; ++b) {
    bOff[b * E_ + e] = run;
    run += bCnt[b * E_ + e];
  }
}

__global__ __launch_bounds__(256)
void init_slots_kernel(int* __restrict__ slot_tok, float* __restrict__ slot_w) {
  const int i = blockIdx.x * 256 + threadIdx.x;
  if (i < EC_) { slot_tok[i] = 0; slot_w[i] = 0.0f; }
}

__global__ __launch_bounds__(256)
void fill_slots_kernel(const int* __restrict__ topk_idx,
                       const float* __restrict__ topk_w,
                       const int* __restrict__ bOff,
                       int* __restrict__ slot_tok,
                       float* __restrict__ slot_w) {
  __shared__ int se[256];
  const int i = blockIdx.x * 256 + threadIdx.x;
  const int e = topk_idx[i];
  se[threadIdx.x] = e;
  __syncthreads();
  int lr = 0;
  for (int j = 0; j < (int)threadIdx.x; ++j) lr += (se[j] == e);
  const int rank = bOff[blockIdx.x * E_ + e] + lr;
  if (rank < C_) {
    slot_tok[e * C_ + rank] = i >> 2;
    slot_w[e * C_ + rank]   = topk_w[i];
  }
}

// =====================================================================
// 4) Routed gate/up GEMM + SiLU*up -> H (f16)
//    block 256 thr (8 waves), tile 32(M)x128(N), K-step 64
//    wave (wm = w>>2, wn = w&3) owns 16x32 -> 8 WMMA / K-step
//    grid = (I_/128, C_/32, E_)
// =====================================================================
__global__ __launch_bounds__(256)
void moe_gateup_kernel(const float* __restrict__ x,
                       const float* __restrict__ Wg,
                       const float* __restrict__ Wu,
                       const int* __restrict__ slot_tok,
                       _Float16* __restrict__ Hro) {
  const int e      = blockIdx.z;
  const int mBase  = blockIdx.y * 32;
  const int nBase0 = blockIdx.x * 128;
  const int tid  = threadIdx.x;
  const int w    = tid >> 5;
  const int wm   = w >> 2;          // 0..1
  const int wn   = w & 3;           // 0..3
  const int lane = tid & 31;
  const int half = lane >> 4;
  const int r    = lane & 15;
  const int kOff = half * 8;

  __shared__ _Float16 As[32 * 64];      // [m][k]
  __shared__ _Float16 Bg[128 * 64];     // transposed [n][k]
  __shared__ _Float16 Bu[128 * 64];
  __shared__ int toks[32];

  if (tid < 32) toks[tid] = slot_tok[e * C_ + mBase + tid];
  __syncthreads();

  const float* wgBase = Wg + (size_t)e * D_ * I_ + nBase0;
  const float* wuBase = Wu + (size_t)e * D_ * I_ + nBase0;

  v8f accG0 = {}, accG1 = {}, accU0 = {}, accU1 = {};

  for (int k0 = 0; k0 < D_; k0 += 64) {
    // stage A: 32x64 fp32 -> f16 (gathered rows), 8 floats/thread, b128 store
    {
      const int row = tid >> 3;            // 0..31
      const int col = (tid & 7) * 8;       // 0..56
      const float* src = x + (size_t)toks[row] * D_ + k0 + col;
      float4 a = *(const float4*)(src);
      float4 b = *(const float4*)(src + 4);
      store8h(As + row * 64 + col, a, b);
    }
    // stage B gate/up: 64(K)x128(N) fp32, float4 along N, transpose to [n][k]
#pragma unroll
    for (int i = 0; i < 8; ++i) {
      const int idx4 = tid + i * 256;      // 0..2047  (float4 index)
      const int kk = idx4 >> 5;            // 0..63
      const int nn = (idx4 & 31) * 4;      // 0..124
      const size_t go = (size_t)(k0 + kk) * I_ + nn;
      float4 g = *(const float4*)(wgBase + go);
      float4 u = *(const float4*)(wuBase + go);
      Bg[(nn + 0) * 64 + kk] = (_Float16)g.x;
      Bg[(nn + 1) * 64 + kk] = (_Float16)g.y;
      Bg[(nn + 2) * 64 + kk] = (_Float16)g.z;
      Bg[(nn + 3) * 64 + kk] = (_Float16)g.w;
      Bu[(nn + 0) * 64 + kk] = (_Float16)u.x;
      Bu[(nn + 1) * 64 + kk] = (_Float16)u.y;
      Bu[(nn + 2) * 64 + kk] = (_Float16)u.z;
      Bu[(nn + 3) * 64 + kk] = (_Float16)u.w;
    }
    if (k0 + 64 < D_)
      __builtin_prefetch(&wgBase[(size_t)(k0 + 64 + (tid >> 5)) * I_ + (tid & 31) * 4], 0, 0);
    __syncthreads();
#pragma unroll
    for (int kk = 0; kk < 64; kk += 32) {
      v16h a = lds_fragA(As + (wm * 16 + r) * 64 + kk, kOff);
      const int c0 = (wn * 32 + r) * 64 + kk;
      const int c1 = (wn * 32 + 16 + r) * 64 + kk;
      accG0 = wmma32(a, lds_fragB(Bg + c0, half), accG0);
      accG1 = wmma32(a, lds_fragB(Bg + c1, half), accG1);
      accU0 = wmma32(a, lds_fragB(Bu + c0, half), accU0);
      accU1 = wmma32(a, lds_fragB(Bu + c1, half), accU1);
    }
    __syncthreads();
  }
  // epilogue: H = silu(G) * U  -> f16
#pragma unroll
  for (int j = 0; j < 8; ++j) {
    const int row = wm * 16 + j + half * 8;
    const size_t base = (size_t)(e * C_ + mBase + row) * I_ + nBase0 + wn * 32;
    Hro[base + r]      = (_Float16)(silu_f(accG0[j]) * accU0[j]);
    Hro[base + 16 + r] = (_Float16)(silu_f(accG1[j]) * accU1[j]);
  }
}

// =====================================================================
// 5) Routed down GEMM, scale by slot weight, atomicAdd to out
//    block 256 thr, tile 32x128, K-step 64; grid = (D_/128, C_/32, E_)
// =====================================================================
__global__ __launch_bounds__(256)
void moe_down_kernel(const _Float16* __restrict__ Hro,
                     const float* __restrict__ Wd,
                     const int* __restrict__ slot_tok,
                     const float* __restrict__ slot_w,
                     float* __restrict__ out) {
  const int e      = blockIdx.z;
  const int mBase  = blockIdx.y * 32;
  const int nBase0 = blockIdx.x * 128;
  const int tid  = threadIdx.x;
  const int w    = tid >> 5;
  const int wm   = w >> 2;
  const int wn   = w & 3;
  const int lane = tid & 31;
  const int half = lane >> 4;
  const int r    = lane & 15;
  const int kOff = half * 8;

  __shared__ _Float16 As[32 * 64];
  __shared__ _Float16 Bt[128 * 64];
  __shared__ int   toks[32];
  __shared__ float wts[32];

  if (tid < 32) {
    toks[tid] = slot_tok[e * C_ + mBase + tid];
    wts[tid]  = slot_w[e * C_ + mBase + tid];
  }
  __syncthreads();

  const float* wdBase = Wd + (size_t)e * I_ * D_ + nBase0;

  v8f acc0 = {}, acc1 = {};

  for (int k0 = 0; k0 < I_; k0 += 64) {
    {   // A already f16: 16B copy per thread
      const int row = tid >> 3;
      const int col = (tid & 7) * 8;
      *(uint4*)(As + row * 64 + col) =
          *(const uint4*)(Hro + (size_t)(e * C_ + mBase + row) * I_ + k0 + col);
    }
#pragma unroll
    for (int i = 0; i < 8; ++i) {
      const int idx4 = tid + i * 256;
      const int kk = idx4 >> 5;
      const int nn = (idx4 & 31) * 4;
      float4 d = *(const float4*)(wdBase + (size_t)(k0 + kk) * D_ + nn);
      Bt[(nn + 0) * 64 + kk] = (_Float16)d.x;
      Bt[(nn + 1) * 64 + kk] = (_Float16)d.y;
      Bt[(nn + 2) * 64 + kk] = (_Float16)d.z;
      Bt[(nn + 3) * 64 + kk] = (_Float16)d.w;
    }
    if (k0 + 64 < I_)
      __builtin_prefetch(&wdBase[(size_t)(k0 + 64 + (tid >> 5)) * D_ + (tid & 31) * 4], 0, 0);
    __syncthreads();
#pragma unroll
    for (int kk = 0; kk < 64; kk += 32) {
      v16h a = lds_fragA(As + (wm * 16 + r) * 64 + kk, kOff);
      acc0 = wmma32(a, lds_fragB(Bt + (wn * 32 + r) * 64 + kk, half), acc0);
      acc1 = wmma32(a, lds_fragB(Bt + (wn * 32 + 16 + r) * 64 + kk, half), acc1);
    }
    __syncthreads();
  }
#pragma unroll
  for (int j = 0; j < 8; ++j) {
    const int row = wm * 16 + j + half * 8;
    const float wt = wts[row];
    float* dst = out + (size_t)toks[row] * D_ + nBase0 + wn * 32;
    atomicAdd(dst + r,      acc0[j] * wt);
    atomicAdd(dst + 16 + r, acc1[j] * wt);
  }
}

// =====================================================================
// 6) Shared expert gate/up: Hs = silu(x@sg^T) * (x@su^T)
//    sg/su are [N][K]-major -> B fragments direct from global.
//    block 256 thr, tile 32x128, K-step 64; grid = (ISH_/128, T_/32)
// =====================================================================
__global__ __launch_bounds__(256)
void shared_gateup_kernel(const float* __restrict__ x,
                          const float* __restrict__ sg,
                          const float* __restrict__ su,
                          _Float16* __restrict__ Hs) {
  const int mBase  = blockIdx.y * 32;
  const int nBase0 = blockIdx.x * 128;
  const int tid  = threadIdx.x;
  const int w    = tid >> 5;
  const int wm   = w >> 2;
  const int wn   = w & 3;
  const int lane = tid & 31;
  const int half = lane >> 4;
  const int r    = lane & 15;
  const int kOff = half * 8;

  __shared__ _Float16 As[32 * 64];

  const int n0 = nBase0 + wn * 32 + r;
  const float* sgRow0 = sg + (size_t)n0 * D_;
  const float* sgRow1 = sgRow0 + (size_t)16 * D_;
  const float* suRow0 = su + (size_t)n0 * D_;
  const float* suRow1 = suRow0 + (size_t)16 * D_;

  v8f accG0 = {}, accG1 = {}, accU0 = {}, accU1 = {};

  for (int k0 = 0; k0 < D_; k0 += 64) {
    {
      const int row = tid >> 3;
      const int col = (tid & 7) * 8;
      const float* src = x + (size_t)(mBase + row) * D_ + k0 + col;
      float4 a = *(const float4*)(src);
      float4 b = *(const float4*)(src + 4);
      store8h(As + row * 64 + col, a, b);
    }
    __syncthreads();
#pragma unroll
    for (int kk = 0; kk < 64; kk += 32) {
      v16h a = lds_fragA(As + (wm * 16 + r) * 64 + kk, kOff);
      accG0 = wmma32(a, glb_fragB_f32(sgRow0 + k0 + kk, half), accG0);
      accG1 = wmma32(a, glb_fragB_f32(sgRow1 + k0 + kk, half), accG1);
      accU0 = wmma32(a, glb_fragB_f32(suRow0 + k0 + kk, half), accU0);
      accU1 = wmma32(a, glb_fragB_f32(suRow1 + k0 + kk, half), accU1);
    }
    __syncthreads();
  }
#pragma unroll
  for (int j = 0; j < 8; ++j) {
    const int row = wm * 16 + j + half * 8;
    const size_t base = (size_t)(mBase + row) * ISH_ + nBase0 + wn * 32;
    Hs[base + r]      = (_Float16)(silu_f(accG0[j]) * accU0[j]);
    Hs[base + 16 + r] = (_Float16)(silu_f(accG1[j]) * accU1[j]);
  }
}

// =====================================================================
// 7) Shared expert down: out = Hs @ sd^T  (dense store: initializes out)
//    grid = (D_/128, T_/32)
// =====================================================================
__global__ __launch_bounds__(256)
void shared_down_kernel(const _Float16* __restrict__ Hs,
                        const float* __restrict__ sd,
                        float* __restrict__ out) {
  const int mBase  = blockIdx.y * 32;
  const int nBase0 = blockIdx.x * 128;
  const int tid  = threadIdx.x;
  const int w    = tid >> 5;
  const int wm   = w >> 2;
  const int wn   = w & 3;
  const int lane = tid & 31;
  const int half = lane >> 4;
  const int r    = lane & 15;
  const int kOff = half * 8;

  __shared__ _Float16 As[32 * 64];

  const int n0 = nBase0 + wn * 32 + r;
  const float* sdRow0 = sd + (size_t)n0 * ISH_;
  const float* sdRow1 = sdRow0 + (size_t)16 * ISH_;

  v8f acc0 = {}, acc1 = {};

  for (int k0 = 0; k0 < ISH_; k0 += 64) {
    {
      const int row = tid >> 3;
      const int col = (tid & 7) * 8;
      *(uint4*)(As + row * 64 + col) =
          *(const uint4*)(Hs + (size_t)(mBase + row) * ISH_ + k0 + col);
    }
    __syncthreads();
#pragma unroll
    for (int kk = 0; kk < 64; kk += 32) {
      v16h a = lds_fragA(As + (wm * 16 + r) * 64 + kk, kOff);
      acc0 = wmma32(a, glb_fragB_f32(sdRow0 + k0 + kk, half), acc0);
      acc1 = wmma32(a, glb_fragB_f32(sdRow1 + k0 + kk, half), acc1);
    }
    __syncthreads();
  }
#pragma unroll
  for (int j = 0; j < 8; ++j) {
    const int row = wm * 16 + j + half * 8;
    float* dst = out + (size_t)(mBase + row) * D_ + nBase0 + wn * 32;
    dst[r]      = acc0[j];
    dst[16 + r] = acc1[j];
  }
}

// =====================================================================
// host launch
// =====================================================================
extern "C" void kernel_launch(void* const* d_in, const int* in_sizes, int n_in,
                              void* d_out, int out_size, void* d_ws, size_t ws_size,
                              hipStream_t stream) {
  (void)in_sizes; (void)n_in; (void)out_size; (void)ws_size;

  const float* x  = (const float*)d_in[0];   // [T, D]
  const float* rw = (const float*)d_in[1];   // [E, D]
  const float* eb = (const float*)d_in[2];   // [E]
  const float* Wg = (const float*)d_in[3];   // [E, D, I]
  const float* Wu = (const float*)d_in[4];   // [E, D, I]
  const float* Wd = (const float*)d_in[5];   // [E, I, D]
  const float* sg = (const float*)d_in[6];   // [ISH, D]
  const float* su = (const float*)d_in[7];   // [ISH, D]
  const float* sd = (const float*)d_in[8];   // [D, ISH]
  float* out = (float*)d_out;                // [T, D]

  char* p = (char*)d_ws;
  auto take = [&](size_t bytes) {
    void* r = (void*)p;
    p += (bytes + 255) & ~(size_t)255;
    return r;
  };
  _Float16* Hs     = (_Float16*)take((size_t)T_ * ISH_ * sizeof(_Float16));
  _Float16* Hro    = (_Float16*)take((size_t)EC_ * I_ * sizeof(_Float16));
  float* scores    = (float*)take((size_t)T_ * E_ * sizeof(float));
  int*   topk_idx  = (int*)take((size_t)T_ * K_ * sizeof(int));
  float* topk_w    = (float*)take((size_t)T_ * K_ * sizeof(float));
  int*   bCnt      = (int*)take((size_t)64 * E_ * sizeof(int));
  int*   bOff      = (int*)take((size_t)64 * E_ * sizeof(int));
  int*   slot_tok  = (int*)take((size_t)EC_ * sizeof(int));
  float* slot_w    = (float*)take((size_t)EC_ * sizeof(float));

  // --- routing ---
  router_scores_kernel<<<T_, 32, 0, stream>>>(x, rw, scores);
  topk_kernel<<<T_ / 256, 256, 0, stream>>>(scores, eb, topk_idx, topk_w);

  // --- dispatch ---
  count_kernel<<<(T_ * K_) / 256, 256, 0, stream>>>(topk_idx, bCnt);
  scan_kernel<<<1, 32, 0, stream>>>(bCnt, bOff);
  init_slots_kernel<<<(EC_ + 255) / 256, 256, 0, stream>>>(slot_tok, slot_w);
  fill_slots_kernel<<<(T_ * K_) / 256, 256, 0, stream>>>(topk_idx, topk_w, bOff,
                                                         slot_tok, slot_w);

  // --- routed experts gate/up (WMMA) ---
  moe_gateup_kernel<<<dim3(I_ / 128, C_ / 32, E_), 256, 0, stream>>>(
      x, Wg, Wu, slot_tok, Hro);

  // --- shared expert gate/up (WMMA) ---
  shared_gateup_kernel<<<dim3(ISH_ / 128, T_ / 32), 256, 0, stream>>>(x, sg, su, Hs);

  // --- shared expert down: writes (initializes) out ---
  shared_down_kernel<<<dim3(D_ / 128, T_ / 32), 256, 0, stream>>>(Hs, sd, out);

  // --- routed experts down: atomic accumulate into out ---
  moe_down_kernel<<<dim3(D_ / 128, C_ / 32, E_), 256, 0, stream>>>(
      Hro, Wd, slot_tok, slot_w, out);
}
